// GraphNetwork_45286135169038
// MI455X (gfx1250) — compile-verified, matrix operand order checked by reading, not run on hardware
//
#include <hip/hip_runtime.h>
#include <hip/hip_bf16.h>
#include <math.h>

// ---------------------------------------------------------------------------
// GraphNetwork (GNS) forward for MI455X / gfx1250.
//
// Each MLP layer is T' = W^T * T + b over a 16-sample tile, done as one
// V_WMMA_F32_16X16X32_BF16 (K padded 16->32). The f32 D layout equals the
// bf16 B layout of the next layer, so chained layers need no lane shuffles.
//
// Optimizations:
//  * Weights pre-packed ONCE PER BLOCK into LDS in bf16 A-layout:
//    per layer each lane needs a single ds_load_b128 (vs 8 global loads).
//  * Biases / LN gammas / betas staged in LDS (broadcast ds reads).
//  * TWO tiles per wave: two independent accumulator chains share each A
//    operand; paired WMMAs fill D->VALU hazard slots.
//  * Layer loops fully unrolled so the scheduler can prefetch next-layer
//    ds loads under the current WMMAs and fill hazard NOPs.
// ---------------------------------------------------------------------------

typedef __attribute__((ext_vector_type(8)))  float  v8f;
typedef __attribute__((ext_vector_type(16))) __bf16 v16bf;

constexpr int B_ = 4;
constexpr int N_ = 16384;
constexpr int E_ = 131072;
constexpr float EPS_ = 1e-5f;

struct MLPp {
  const float *W_in, *W_out, *Wh, *b_in, *b_out, *be0, *beh, *bh, *g0, *gh;
};

__device__ inline v8f wmma_bf16(v16bf A, v16bf B, v8f C) {
  return __builtin_amdgcn_wmma_f32_16x16x32_bf16(false, A, false, B,
                                                 (short)0, C, false, false);
}

union APk { v16bf v; uint32_t u[8]; };

// A operand for one layer: one ds_load_b128 of the pre-packed bf16 weights.
__device__ inline v16bf load_A(const uint4* __restrict__ wl, int lane) {
  APk a;
  uint4 w = wl[lane];
  a.u[0] = w.x; a.u[1] = w.y; a.u[2] = w.z; a.u[3] = w.w;
  a.u[4] = 0; a.u[5] = 0; a.u[6] = 0; a.u[7] = 0;
  return a.v;
}

__device__ inline v16bf cvt_B(const v8f x) {
  v16bf b = {};
#pragma unroll
  for (int j = 0; j < 8; ++j) b[j] = (__bf16)x[j];
  return b;
}

// One hidden layer applied to two independent tiles (shared A, shared bias).
__device__ inline void layer2(v8f& x0, v8f& x1, const uint4* __restrict__ wl,
                              const float* __restrict__ b, int kbase, int lane) {
  v16bf A = load_A(wl, lane);
  v8f C;
#pragma unroll
  for (int v = 0; v < 8; ++v) C[v] = b[kbase + v];
  v16bf B0 = cvt_B(x0), B1 = cvt_B(x1);
  x0 = wmma_bf16(A, B0, C);
  x1 = wmma_bf16(A, B1, C);
}

// LayerNorm over 16 features of each sample (lane pair s, s+16), f32.
__device__ inline v8f ln16(v8f x, const float* __restrict__ g,
                           const float* __restrict__ be, int kbase) {
  float s = 0.f;
#pragma unroll
  for (int v = 0; v < 8; ++v) s += x[v];
  s += __shfl_xor(s, 16, 32);
  float mu = s * 0.0625f;
  float var = 0.f;
#pragma unroll
  for (int v = 0; v < 8; ++v) { float d = x[v] - mu; var += d * d; }
  var += __shfl_xor(var, 16, 32);
  float rstd = rsqrtf(var * 0.0625f + EPS_);
  v8f y;
#pragma unroll
  for (int v = 0; v < 8; ++v)
    y[v] = (x[v] - mu) * rstd * g[kbase + v] + be[kbase + v];
  return y;
}

// Input layer din->16 in f32 VALU from an LDS-staged sample row.
__device__ inline v8f in_layer(const float* __restrict__ xrow, int din,
                               const float* __restrict__ Win,
                               const float* __restrict__ bin, int kbase) {
  v8f acc;
#pragma unroll
  for (int v = 0; v < 8; ++v) acc[v] = bin[kbase + v];
  for (int k = 0; k < din; ++k) {
    float xk = xrow[k];
#pragma unroll
    for (int v = 0; v < 8; ++v)
      acc[v] = fmaf(xk, Win[k * 16 + kbase + v], acc[v]);
  }
  return acc;
}

// 17 plain layers (16 hidden + 16x16 W_out), dual tile, fully unrolled.
__device__ inline void mlp_plain2(v8f& x0, v8f& x1, const uint4* wpk,
                                  const float* lbh, int kbase, int lane) {
#pragma unroll
  for (int l = 0; l < 17; ++l)
    layer2(x0, x1, wpk + l * 32, lbh + l * 16, kbase, lane);
}

// LN-in, 16x(layer+LN), then 16x16 W_out, dual tile, fully unrolled.
__device__ inline void mlp_bn2(v8f& x0, v8f& x1, const uint4* wpk,
                               const float* lbh, const float* lgh,
                               const float* lbe, int kbase, int lane) {
  x0 = ln16(x0, lgh, lbe, kbase);
  x1 = ln16(x1, lgh, lbe, kbase);
#pragma unroll
  for (int l = 0; l < 16; ++l) {
    layer2(x0, x1, wpk + l * 32, lbh + l * 16, kbase, lane);
    x0 = ln16(x0, lgh + (l + 1) * 16, lbe + (l + 1) * 16, kbase);
    x1 = ln16(x1, lgh + (l + 1) * 16, lbe + (l + 1) * 16, kbase);
  }
  layer2(x0, x1, wpk + 16 * 32, lbh + 16 * 16, kbase, lane);
}

// Cooperative pack: nlayers 16x16 f32 matrices -> bf16 A-layout in LDS.
// dst dword index: layer*128 + lane*4 + j2, holding packed (k=kb+2j2, kb+2j2+1), kb=8*(lane/16).
__device__ inline void pack_layers(const float* __restrict__ W,
                                   uint32_t* __restrict__ dst, int nlayers) {
  const int total = nlayers * 128;
  for (int i = threadIdx.x; i < total; i += 256) {
    const int l = i >> 7, r = i & 127;
    const int p = r >> 2, j2 = r & 3;
    const int col = p & 15, kb = (p >> 4) * 8;
    const float* Wl = W + l * 256;
    union { __bf16 h[2]; uint32_t u; } pk;
    pk.h[0] = (__bf16)Wl[(kb + 2 * j2) * 16 + col];
    pk.h[1] = (__bf16)Wl[(kb + 2 * j2 + 1) * 16 + col];
    dst[i] = pk.u;
  }
}

__device__ inline void copy_f(const float* __restrict__ s,
                              float* __restrict__ d, int n) {
  for (int i = threadIdx.x; i < n; i += 256) d[i] = s[i];
}

// ---------------------------------------------------------------------------

__global__ __launch_bounds__(256) void k_zero(float* p, int n) {
  int i = blockIdx.x * blockDim.x + threadIdx.x;
  if (i < n) p[i] = 0.f;
}

__global__ __launch_bounds__(256)
void k_node_enc(const float* __restrict__ nodes, const float* __restrict__ glob,
                MLPp p, float* __restrict__ ve) {
  __shared__ uint4 wpk[17 * 32];
  __shared__ float lbh[17 * 16];
  __shared__ float xt[8][2][16 * 7];
  pack_layers(p.Wh, (uint32_t*)wpk, 16);
  pack_layers(p.W_out, (uint32_t*)(wpk + 16 * 32), 1);
  copy_f(p.bh, lbh, 256);
  copy_f(p.b_out, lbh + 256, 16);
  const int lane = threadIdx.x & 31, wv = threadIdx.x >> 5;
  const int s = lane & 15, half = lane >> 4, kbase = half * 8;
  const long base = (long)(blockIdx.x * 8 + wv) * 32;  // 2 tiles = 32 rows
  {
    const long row = base + half * 16 + s;
    const float* nr = nodes + row * 6;
    float* xr = &xt[wv][half][s * 7];
#pragma unroll
    for (int k = 0; k < 6; ++k) xr[k] = nr[k];
    xr[6] = glob[(int)(row / N_)];
  }
  __syncthreads();
  v8f x0 = in_layer(&xt[wv][0][s * 7], 7, p.W_in, p.b_in, kbase);
  v8f x1 = in_layer(&xt[wv][1][s * 7], 7, p.W_in, p.b_in, kbase);
  mlp_plain2(x0, x1, wpk, lbh, kbase, lane);
  float* o0 = ve + (base + s) * 16 + kbase;
  float* o1 = ve + (base + 16 + s) * 16 + kbase;
#pragma unroll
  for (int v = 0; v < 8; ++v) { o0[v] = x0[v]; o1[v] = x1[v]; }
}

__global__ __launch_bounds__(256)
void k_edge_enc(const float* __restrict__ edges, const float* __restrict__ nodes,
                const float* __restrict__ glob, const int* __restrict__ senders,
                const int* __restrict__ receivers, MLPp p,
                float* __restrict__ ee) {
  __shared__ uint4 wpk[17 * 32];
  __shared__ float lbh[17 * 16];
  __shared__ float xt[8][2][16 * 5];
  pack_layers(p.Wh, (uint32_t*)wpk, 16);
  pack_layers(p.W_out, (uint32_t*)(wpk + 16 * 32), 1);
  copy_f(p.bh, lbh, 256);
  copy_f(p.b_out, lbh + 256, 16);
  const int lane = threadIdx.x & 31, wv = threadIdx.x >> 5;
  const int s = lane & 15, half = lane >> 4, kbase = half * 8;
  const long base = (long)(blockIdx.x * 8 + wv) * 32;
  {
    const long row = base + half * 16 + s;
    const int b = (int)(row / E_);
    const long nb = (long)b * N_;
    const float* ps = nodes + (nb + senders[row]) * 6;
    const float* pr = nodes + (nb + receivers[row]) * 6;
    float r0 = ps[0] - pr[0], r1 = ps[1] - pr[1], r2 = ps[2] - pr[2];
    float* xr = &xt[wv][half][s * 5];
    xr[0] = edges[row];
    xr[1] = r0; xr[2] = r1; xr[3] = r2;
    xr[4] = sqrtf(r0 * r0 + r1 * r1 + r2 * r2);
  }
  __syncthreads();
  v8f x0 = in_layer(&xt[wv][0][s * 5], 5, p.W_in, p.b_in, kbase);
  v8f x1 = in_layer(&xt[wv][1][s * 5], 5, p.W_in, p.b_in, kbase);
  mlp_plain2(x0, x1, wpk, lbh, kbase, lane);
  float* o0 = ee + (base + s) * 16 + kbase;
  float* o1 = ee + (base + 16 + s) * 16 + kbase;
#pragma unroll
  for (int v = 0; v < 8; ++v) { o0[v] = x0[v]; o1[v] = x1[v]; }
}

__global__ __launch_bounds__(256)
void k_edge_proc(const float* __restrict__ eein, const float* __restrict__ ve,
                 const float* __restrict__ glob, const int* __restrict__ senders,
                 const int* __restrict__ receivers, MLPp p,
                 float* __restrict__ agg) {
  __shared__ uint4 wpk[17 * 32];
  __shared__ float lbh[17 * 16], lgh[17 * 16], lbe[17 * 16];
  __shared__ float xt[8][2][16 * 49];
  pack_layers(p.Wh, (uint32_t*)wpk, 16);
  pack_layers(p.W_out, (uint32_t*)(wpk + 16 * 32), 1);
  copy_f(p.bh, lbh, 256);  copy_f(p.b_out, lbh + 256, 16);
  copy_f(p.g0, lgh, 16);   copy_f(p.gh, lgh + 16, 256);
  copy_f(p.be0, lbe, 16);  copy_f(p.beh, lbe + 16, 256);
  const int lane = threadIdx.x & 31, wv = threadIdx.x >> 5;
  const int s = lane & 15, half = lane >> 4, kbase = half * 8;
  const long base = (long)(blockIdx.x * 8 + wv) * 32;
  const int b = (int)(base / E_);       // base % 32 == 0, batch edge-aligned
  const long nb = (long)b * N_;
  {
    const long row = base + half * 16 + s;
    const float* er = eein + row * 16;
    const float* vs = ve + (nb + senders[row]) * 16;
    const float* vr = ve + (nb + receivers[row]) * 16;
    float* xr = &xt[wv][half][s * 49];
#pragma unroll
    for (int k = 0; k < 16; ++k) {
      xr[k] = er[k]; xr[16 + k] = vs[k]; xr[32 + k] = vr[k];
    }
    xr[48] = glob[b];
  }
  __syncthreads();
  v8f x0 = in_layer(&xt[wv][0][s * 49], 49, p.W_in, p.b_in, kbase);
  v8f x1 = in_layer(&xt[wv][1][s * 49], 49, p.W_in, p.b_in, kbase);
  mlp_bn2(x0, x1, wpk, lbh, lgh, lbe, kbase, lane);
  // segment_sum: scatter-add both tiles' rows into agg[receiver].
  float* a0 = agg + (nb + receivers[base + s]) * 16 + kbase;
  float* a1 = agg + (nb + receivers[base + 16 + s]) * 16 + kbase;
#pragma unroll
  for (int v = 0; v < 8; ++v) { atomicAdd(a0 + v, x0[v]); atomicAdd(a1 + v, x1[v]); }
}

__global__ __launch_bounds__(256)
void k_node_dec(const float* __restrict__ agg, const float* __restrict__ vein,
                const float* __restrict__ glob, MLPp pn, MLPp pd,
                float* __restrict__ out) {
  __shared__ uint4 wpk[34 * 32];  // 0..15 nproc Wh, 16 nproc Wout,
                                  // 17 dec Win, 18..33 dec Wh
  __shared__ float lbh[17 * 16], lgh[17 * 16], lbe[17 * 16];
  __shared__ float ldb[17 * 16];  // dec b_in (slot 0) + dec bh (slots 1..16)
  __shared__ float xt[8][2][16 * 33];
  pack_layers(pn.Wh, (uint32_t*)wpk, 16);
  pack_layers(pn.W_out, (uint32_t*)(wpk + 16 * 32), 1);
  pack_layers(pd.W_in, (uint32_t*)(wpk + 17 * 32), 1);
  pack_layers(pd.Wh, (uint32_t*)(wpk + 18 * 32), 16);
  copy_f(pn.bh, lbh, 256);  copy_f(pn.b_out, lbh + 256, 16);
  copy_f(pn.g0, lgh, 16);   copy_f(pn.gh, lgh + 16, 256);
  copy_f(pn.be0, lbe, 16);  copy_f(pn.beh, lbe + 16, 256);
  copy_f(pd.b_in, ldb, 16); copy_f(pd.bh, ldb + 16, 256);
  const int lane = threadIdx.x & 31, wv = threadIdx.x >> 5;
  const int s = lane & 15, half = lane >> 4, kbase = half * 8;
  const long base = (long)(blockIdx.x * 8 + wv) * 32;
  {
    const long row = base + half * 16 + s;
    const float* ar = agg + row * 16;
    const float* vr = vein + row * 16;
    float* xr = &xt[wv][half][s * 33];
#pragma unroll
    for (int k = 0; k < 16; ++k) { xr[k] = ar[k]; xr[16 + k] = vr[k]; }
    xr[32] = glob[(int)(row / N_)];
  }
  __syncthreads();
  v8f x0 = in_layer(&xt[wv][0][s * 33], 33, pn.W_in, pn.b_in, kbase);
  v8f x1 = in_layer(&xt[wv][1][s * 33], 33, pn.W_in, pn.b_in, kbase);
  mlp_bn2(x0, x1, wpk, lbh, lgh, lbe, kbase, lane);  // -> new ve in regs
  // dec: W_in + 16 hidden layers, all 16x16 WMMA layers, fully unrolled.
#pragma unroll
  for (int l = 0; l < 17; ++l)
    layer2(x0, x1, wpk + (17 + l) * 32, ldb + l * 16, kbase, lane);
  // dec output layer 16 -> 3 (A rows >= 3 zero), straight from global.
  v16bf A = {};
  if (s < 3) {
#pragma unroll
    for (int j = 0; j < 8; ++j) A[j] = (__bf16)pd.W_out[(kbase + j) * 3 + s];
  }
  v8f C;
#pragma unroll
  for (int v = 0; v < 8; ++v)
    C[v] = (kbase + v < 3) ? pd.b_out[kbase + v] : 0.f;
  v8f D0 = wmma_bf16(A, cvt_B(x0), C);
  v8f D1 = wmma_bf16(A, cvt_B(x1), C);
  if (lane < 16) {  // lane s holds out[row, 0..2] in D[0..2]
    float* o0 = out + (base + s) * 3;
    float* o1 = out + (base + 16 + s) * 3;
    o0[0] = D0[0]; o0[1] = D0[1]; o0[2] = D0[2];
    o1[0] = D1[0]; o1[1] = D1[1]; o1[2] = D1[2];
  }
}

// ---------------------------------------------------------------------------

extern "C" void kernel_launch(void* const* d_in, const int* in_sizes, int n_in,
                              void* d_out, int out_size, void* d_ws,
                              size_t ws_size, hipStream_t stream) {
  (void)in_sizes; (void)n_in; (void)out_size; (void)ws_size;
  const float* nodes     = (const float*)d_in[0];
  const float* edges     = (const float*)d_in[1];
  const float* glob      = (const float*)d_in[2];
  const int*   senders   = (const int*)d_in[3];
  const int*   receivers = (const int*)d_in[4];

  // params flattened with sorted dict keys: dec, edge_enc, edge_proc,
  // node_enc, node_proc; within each MLP: W_in, W_out, Wh, b_in, b_out,
  // [be0, beh,] bh [, g0, gh].
  int i = 5;
  auto take = [&](MLPp& m, bool bn) {
    m.W_in  = (const float*)d_in[i++];
    m.W_out = (const float*)d_in[i++];
    m.Wh    = (const float*)d_in[i++];
    m.b_in  = (const float*)d_in[i++];
    m.b_out = (const float*)d_in[i++];
    if (bn) {
      m.be0 = (const float*)d_in[i++];
      m.beh = (const float*)d_in[i++];
      m.bh  = (const float*)d_in[i++];
      m.g0  = (const float*)d_in[i++];
      m.gh  = (const float*)d_in[i++];
    } else {
      m.bh = (const float*)d_in[i++];
      m.be0 = m.beh = m.g0 = m.gh = nullptr;
    }
  };
  MLPp p_dec, p_eenc, p_eproc, p_nenc, p_nproc;
  take(p_dec, false);
  take(p_eenc, false);
  take(p_eproc, true);
  take(p_nenc, false);
  take(p_nproc, true);

  // Workspace carve-out: ve (4MB) | agg (4MB) | ee (32MB).
  float* ve  = (float*)d_ws;
  float* agg = ve + (size_t)B_ * N_ * 16;
  float* eeb = agg + (size_t)B_ * N_ * 16;

  const int nagg = B_ * N_ * 16;
  const int node_blocks = (B_ * N_) / 256;  // 256 rows per block (2 tiles/wave)
  const int edge_blocks = (B_ * E_) / 256;  // 2048

  k_zero<<<(nagg + 255) / 256, 256, 0, stream>>>(agg, nagg);
  k_node_enc<<<node_blocks, 256, 0, stream>>>(nodes, glob, p_nenc, ve);
  k_edge_enc<<<edge_blocks, 256, 0, stream>>>(edges, nodes, glob, senders,
                                              receivers, p_eenc, eeb);
  k_edge_proc<<<edge_blocks, 256, 0, stream>>>(eeb, ve, glob, senders,
                                               receivers, p_eproc, agg);
  k_node_dec<<<node_blocks, 256, 0, stream>>>(agg, ve, glob, p_nproc, p_dec,
                                              (float*)d_out);
}
